// GCNLinkPredictor_61435212202470
// MI455X (gfx1250) — matrix operand order repaired and checked by reading, not compiled, not run on hardware
//
#include <hip/hip_runtime.h>
#include <stdint.h>

typedef __bf16 bf16;
typedef __attribute__((ext_vector_type(16))) __bf16 v16bf;
typedef __attribute__((ext_vector_type(8)))  __bf16 v8bf;
typedef __attribute__((ext_vector_type(8)))  float  v8f;

#define N_NODESC 100000
#define IN_DIMC  128
#define HIDC     256
#define N_EDGESC 640000
#define N_PAIRSC 262144
#define PAIR_DIMC 768

#define BM 128
#define BN 64
#define BK 32
#define LDA  40   // LDS K-stride (bf16) for A tile  (80 B: 16B-aligned rows, 20-bank stride)
#define LDBT 40   // LDS K-stride (bf16) for B^T tile

__device__ __forceinline__ bf16 f2bf(float f) {
  uint32_t u = __builtin_bit_cast(uint32_t, f);
  u = (u + 0x7FFFu + ((u >> 16) & 1u)) >> 16;   // round-to-nearest-even
  return __builtin_bit_cast(bf16, (uint16_t)u);
}

__device__ __forceinline__ uint32_t pack2(float lo, float hi) {
  return (uint32_t)__builtin_bit_cast(uint16_t, f2bf(lo)) |
         ((uint32_t)__builtin_bit_cast(uint16_t, f2bf(hi)) << 16);
}

__device__ __forceinline__ v16bf cat8(v8bf lo, v8bf hi) {
  return __builtin_shufflevector(lo, hi, 0, 1, 2, 3, 4, 5, 6, 7,
                                 8, 9, 10, 11, 12, 13, 14, 15);
}

// One K-step of WMMA for a wave owning a 32x32 output sub-tile (2x2 of 16x16).
// A tile: row-major [BM][LDA]; B tile: transposed [BN][LDBT] so every lane's
// 16 fragment values are two aligned 16B LDS chunks (ds_load_b128).
__device__ __forceinline__ void wave_step(const bf16* __restrict__ sA,
                                          const bf16* __restrict__ sBt,
                                          int wm, int wn, int lane,
                                          v8f acc[2][2]) {
  const int hi16 = lane >> 4;
  const int l16  = lane & 15;
  v16bf a[2];
#pragma unroll
  for (int mi = 0; mi < 2; ++mi) {
    const v8bf* pa =
        (const v8bf*)(sA + (size_t)(wm * 32 + mi * 16 + l16) * LDA + hi16 * 8);
    a[mi] = cat8(pa[0], pa[2]);   // K = hi16*8..+7  and  16+hi16*8..+7
  }
#pragma unroll
  for (int ni = 0; ni < 2; ++ni) {
    const v8bf* pb =
        (const v8bf*)(sBt + (size_t)(wn * 32 + ni * 16 + l16) * LDBT + hi16 * 8);
    const v16bf b = cat8(pb[0], pb[2]);  // K = hi16*8..+7  and  16+hi16*8..+7
#pragma unroll
    for (int mi = 0; mi < 2; ++mi)
      acc[mi][ni] = __builtin_amdgcn_wmma_f32_16x16x32_bf16(
          false, a[mi], false, b, (short)0, acc[mi][ni], false, false);
  }
}

// Generic bf16 GEMM: C[M,N] = A[M,K] * Bt[N,K]^T (+bias, relu); f32 or bf16 out.
// Requires: K % 32 == 0 (>= 64), N % 64 == 0. M guarded (clamped loads).
// Double-buffered LDS, one barrier per K-step, prefetch under WMMA.
__global__ __launch_bounds__(256) void k_gemm(
    const bf16* __restrict__ A, const bf16* __restrict__ Bt,
    float* __restrict__ Cf, bf16* __restrict__ Cb,
    const float* __restrict__ bias, int relu,
    int M, int N, int K) {
  __shared__ bf16 sA[2][BM * LDA];
  __shared__ bf16 sBt[2][BN * LDBT];
  const int tid = threadIdx.x;
  const int bm0 = blockIdx.y * BM;
  const int bn0 = blockIdx.x * BN;
  const int wave = tid >> 5, lane = tid & 31;
  const int wm = wave & 3, wn = wave >> 2;

  auto fetchA = [&](int k0, uint2* r) {
#pragma unroll
    for (int i = 0; i < 4; ++i) {
      const int q = tid + i * 256;
      int grow = bm0 + (q >> 3);
      grow = grow < M ? grow : M - 1;   // clamp: OOB rows computed, never stored
      r[i] = *(const uint2*)(A + (size_t)grow * K + k0 + ((q & 7) << 2));
    }
  };
  auto fetchB = [&](int k0, uint2* r) {
#pragma unroll
    for (int i = 0; i < 2; ++i) {
      const int q = tid + i * 256;
      r[i] = *(const uint2*)(Bt + (size_t)(bn0 + (q >> 3)) * K + k0 + ((q & 7) << 2));
    }
  };
  auto stage = [&](int b, const uint2* ra, const uint2* rb) {
#pragma unroll
    for (int i = 0; i < 4; ++i) {
      const int q = tid + i * 256;
      *(uint2*)(&sA[b][(q >> 3) * LDA + ((q & 7) << 2)]) = ra[i];
    }
#pragma unroll
    for (int i = 0; i < 2; ++i) {
      const int q = tid + i * 256;
      *(uint2*)(&sBt[b][(q >> 3) * LDBT + ((q & 7) << 2)]) = rb[i];
    }
  };

  const v8f vzero = {0.f, 0.f, 0.f, 0.f, 0.f, 0.f, 0.f, 0.f};
  v8f acc[2][2];
  acc[0][0] = vzero; acc[0][1] = vzero; acc[1][0] = vzero; acc[1][1] = vzero;

  const int nstep = K / BK;
  {
    uint2 ra[4], rb[2];
    fetchA(0, ra); fetchB(0, rb);
    stage(0, ra, rb);
  }
  for (int it = 0; it < nstep - 1; ++it) {
    __syncthreads();
    uint2 ra[4], rb[2];
    fetchA((it + 1) * BK, ra);            // prefetch under compute
    fetchB((it + 1) * BK, rb);
    wave_step(sA[it & 1], sBt[it & 1], wm, wn, lane, acc);
    stage((it + 1) & 1, ra, rb);
  }
  __syncthreads();
  wave_step(sA[(nstep - 1) & 1], sBt[(nstep - 1) & 1], wm, wn, lane, acc);

  const int hi16 = lane >> 4, l16 = lane & 15;
  const bool interior = (bm0 + BM) <= M;
#pragma unroll
  for (int mi = 0; mi < 2; ++mi)
#pragma unroll
    for (int ni = 0; ni < 2; ++ni) {
      const int gcol = bn0 + wn * 32 + ni * 16 + l16;
      const float bv = bias ? bias[gcol] : 0.f;
      if (interior) {
#pragma unroll
        for (int r = 0; r < 8; ++r) {
          const int grow = bm0 + wm * 32 + mi * 16 + hi16 * 8 + r;
          float val = acc[mi][ni][r] + bv;
          if (relu) val = fmaxf(val, 0.f);
          if (Cf) Cf[(size_t)grow * N + gcol] = val;
          else    Cb[(size_t)grow * N + gcol] = f2bf(val);
        }
      } else {
#pragma unroll
        for (int r = 0; r < 8; ++r) {
          const int grow = bm0 + wm * 32 + mi * 16 + hi16 * 8 + r;
          if (grow < M) {
            float val = acc[mi][ni][r] + bv;
            if (relu) val = fmaxf(val, 0.f);
            if (Cf) Cf[(size_t)grow * N + gcol] = val;
            else    Cb[(size_t)grow * N + gcol] = f2bf(val);
          }
        }
      }
    }
}

// MLP layer-1 GEMM with on-the-fly pair-feature construction of the A tile:
// feat = [|hu-hv|, hu*hv, |xu-xv|, xu*xv]  (K = 768), fused bias+relu, bf16 out.
__global__ __launch_bounds__(256) void k_gemm_feat(
    const int* __restrict__ pairs, const float* __restrict__ h,
    const float* __restrict__ x, const bf16* __restrict__ Bt,
    bf16* __restrict__ Cb, const float* __restrict__ bias) {
  const int N = HIDC, K = PAIR_DIMC;
  __shared__ bf16 sA[2][BM * LDA];
  __shared__ bf16 sBt[2][BN * LDBT];
  const int tid = threadIdx.x;
  const int bm0 = blockIdx.y * BM;
  const int bn0 = blockIdx.x * BN;
  const int wave = tid >> 5, lane = tid & 31;
  const int wm = wave & 3, wn = wave >> 2;

  const int row = tid >> 1;
  const int c0  = (tid & 1) << 4;
  const int p   = bm0 + row;                 // N_PAIRS % BM == 0
  const int u   = pairs[p];
  const int v   = pairs[N_PAIRSC + p];

  auto buildA = [&](int k0, uint32_t* pk) {  // 16 feature cols -> 8 packed dwords
    const int c = k0 + c0;                   // segment-uniform within 16 cols
    float f[16];
    if (c < 2 * HIDC) {
      const int cc = c & (HIDC - 1);
      const float* pu = h + (size_t)u * HIDC + cc;
      const float* pv = h + (size_t)v * HIDC + cc;
      if (c < HIDC) {
#pragma unroll
        for (int j = 0; j < 16; ++j) f[j] = fabsf(pu[j] - pv[j]);
      } else {
#pragma unroll
        for (int j = 0; j < 16; ++j) f[j] = pu[j] * pv[j];
      }
    } else {
      const int cc = (c - 2 * HIDC) & (IN_DIMC - 1);
      const float* pu = x + (size_t)u * IN_DIMC + cc;
      const float* pv = x + (size_t)v * IN_DIMC + cc;
      if (c < 2 * HIDC + IN_DIMC) {
#pragma unroll
        for (int j = 0; j < 16; ++j) f[j] = fabsf(pu[j] - pv[j]);
      } else {
#pragma unroll
        for (int j = 0; j < 16; ++j) f[j] = pu[j] * pv[j];
      }
    }
#pragma unroll
    for (int j = 0; j < 8; ++j) pk[j] = pack2(f[2 * j], f[2 * j + 1]);
  };
  auto fetchB = [&](int k0, uint2* r) {
#pragma unroll
    for (int i = 0; i < 2; ++i) {
      const int q = tid + i * 256;
      r[i] = *(const uint2*)(Bt + (size_t)(bn0 + (q >> 3)) * K + k0 + ((q & 7) << 2));
    }
  };
  auto stage = [&](int b, const uint32_t* pk, const uint2* rb) {
    uint4* dst = (uint4*)(&sA[b][row * LDA + c0]);     // 2x ds_store_b128
    dst[0] = make_uint4(pk[0], pk[1], pk[2], pk[3]);
    dst[1] = make_uint4(pk[4], pk[5], pk[6], pk[7]);
#pragma unroll
    for (int i = 0; i < 2; ++i) {
      const int q = tid + i * 256;
      *(uint2*)(&sBt[b][(q >> 3) * LDBT + ((q & 7) << 2)]) = rb[i];
    }
  };

  const v8f vzero = {0.f, 0.f, 0.f, 0.f, 0.f, 0.f, 0.f, 0.f};
  v8f acc[2][2];
  acc[0][0] = vzero; acc[0][1] = vzero; acc[1][0] = vzero; acc[1][1] = vzero;

  const int nstep = K / BK;
  {
    uint32_t pk[8]; uint2 rb[2];
    buildA(0, pk); fetchB(0, rb);
    stage(0, pk, rb);
  }
  for (int it = 0; it < nstep - 1; ++it) {
    __syncthreads();
    uint32_t pk[8]; uint2 rb[2];
    buildA((it + 1) * BK, pk);             // gather+compute under WMMA
    fetchB((it + 1) * BK, rb);
    wave_step(sA[it & 1], sBt[it & 1], wm, wn, lane, acc);
    stage((it + 1) & 1, pk, rb);
  }
  __syncthreads();
  wave_step(sA[(nstep - 1) & 1], sBt[(nstep - 1) & 1], wm, wn, lane, acc);

  const int hi16 = lane >> 4, l16 = lane & 15;
#pragma unroll
  for (int mi = 0; mi < 2; ++mi)
#pragma unroll
    for (int ni = 0; ni < 2; ++ni) {
      const int gcol = bn0 + wn * 32 + ni * 16 + l16;
      const float bv = bias[gcol];
#pragma unroll
      for (int r = 0; r < 8; ++r) {
        const int grow = bm0 + wm * 32 + mi * 16 + hi16 * 8 + r;
        Cb[(size_t)grow * N + gcol] = f2bf(fmaxf(acc[mi][ni][r] + bv, 0.f));
      }
    }
}

__global__ void k_f32_to_bf16(const float* __restrict__ in, bf16* __restrict__ out, int n) {
  int t = blockIdx.x * blockDim.x + threadIdx.x;
  if (t < n) out[t] = f2bf(in[t]);
}

// f32 row-major [K,N] -> bf16 transposed [N,K] (one-time weight conversion)
__global__ void k_w_to_bf16t(const float* __restrict__ in, bf16* __restrict__ out,
                             int K, int N) {
  int t = blockIdx.x * blockDim.x + threadIdx.x;
  if (t >= K * N) return;
  int k = t / N, n = t - k * N;
  out[(size_t)n * K + k] = f2bf(in[t]);
}

__global__ void k_deg(const int* __restrict__ dst, float* __restrict__ deg, int E) {
  int t = blockIdx.x * blockDim.x + threadIdx.x;
  if (t < E) atomicAdd(&deg[dst[t]], 1.0f);
}

__global__ void k_dinv(float* __restrict__ deg, int n) {
  int t = blockIdx.x * blockDim.x + threadIdx.x;
  if (t < n) deg[t] = rsqrtf(deg[t] + 1.0f);   // +1 self-loop
}

// out[d] += xw[s] * dinv[s]*dinv[d]  -- 4 features per thread
__global__ void k_edge_scatter(const int* __restrict__ src, const int* __restrict__ dst,
                               const float* __restrict__ dinv, const float* __restrict__ xw,
                               float* __restrict__ agg, int E) {
  int t = blockIdx.x * blockDim.x + threadIdx.x;
  int e = t >> 6;
  if (e >= E) return;
  int f0 = (t & 63) << 2;
  int s = src[e], d = dst[e];
  float nrm = dinv[s] * dinv[d];
  float4 w = *(const float4*)(xw + (size_t)s * HIDC + f0);
  float* o = agg + (size_t)d * HIDC + f0;
  atomicAdd(o + 0, w.x * nrm);
  atomicAdd(o + 1, w.y * nrm);
  atomicAdd(o + 2, w.z * nrm);
  atomicAdd(o + 3, w.w * nrm);
}

// self-loop contribution: agg[i] += xw[i] * dinv[i]^2
__global__ void k_self_add(const float* __restrict__ dinv, const float* __restrict__ xw,
                           float* __restrict__ agg, int n) {
  int t = blockIdx.x * blockDim.x + threadIdx.x;
  if (t >= n) return;
  int i = t >> 6, f0 = (t & 63) << 2;
  float nn = dinv[i] * dinv[i];
  float4 w = *(const float4*)(xw + (size_t)i * HIDC + f0);
  float4* o = (float4*)(agg + (size_t)i * HIDC + f0);
  float4 c = *o;
  c.x += w.x * nn; c.y += w.y * nn; c.z += w.z * nn; c.w += w.w * nn;
  *o = c;
}

// coalesced per-channel partial sums (blockDim == HID)
__global__ void k_colsum(const float* __restrict__ v, float* __restrict__ sums,
                         float* __restrict__ sumsq, int N, int rows_per_block) {
  int c = threadIdx.x;
  int r0 = blockIdx.x * rows_per_block;
  int r1 = min(r0 + rows_per_block, N);
  float s = 0.f, sq = 0.f;
  for (int r = r0; r < r1; ++r) {
    float val = v[(size_t)r * HIDC + c];
    s += val; sq += val * val;
  }
  atomicAdd(&sums[c], s);
  atomicAdd(&sumsq[c], sq);
}

__global__ void k_bn_finalize(const float* __restrict__ sums, const float* __restrict__ sumsq,
                              const float* __restrict__ gamma, const float* __restrict__ beta,
                              float* __restrict__ ss, int N) {
  int c = threadIdx.x;
  float mu = sums[c] / (float)N;
  float var = sumsq[c] / (float)N - mu * mu;
  float sc = gamma[c] * rsqrtf(var + 1e-5f);
  ss[c] = sc;
  ss[HIDC + c] = beta[c] - mu * sc;
}

__global__ void k_bn_relu_bf16(const float* __restrict__ v, const float* __restrict__ ss,
                               bf16* __restrict__ out, int total) {
  int t = blockIdx.x * blockDim.x + threadIdx.x;
  if (t >= total) return;
  int c = t & (HIDC - 1);
  out[t] = f2bf(fmaxf(v[t] * ss[c] + ss[HIDC + c], 0.f));
}

__global__ void k_bn_f32(float* __restrict__ v, const float* __restrict__ ss, int total) {
  int t = blockIdx.x * blockDim.x + threadIdx.x;
  if (t >= total) return;
  int c = t & (HIDC - 1);
  v[t] = v[t] * ss[c] + ss[HIDC + c];
}

__global__ void k_mlp3(const bf16* __restrict__ z2, const float* __restrict__ w3,
                       const float* __restrict__ b3, float* __restrict__ out, int P) {
  int p = blockIdx.x * blockDim.x + threadIdx.x;
  if (p >= P) return;
  const bf16* r = z2 + (size_t)p * 128;
  float s = 0.f;
#pragma unroll 16
  for (int i = 0; i < 128; ++i) s += (float)r[i] * w3[i];
  out[p] = s + b3[0];
}

extern "C" void kernel_launch(void* const* d_in, const int* in_sizes, int n_in,
                              void* d_out, int out_size, void* d_ws, size_t ws_size,
                              hipStream_t stream) {
  (void)in_sizes; (void)n_in; (void)out_size; (void)ws_size;
  const float* x     = (const float*)d_in[0];
  const int*   ei    = (const int*)d_in[1];     // [2,E]: src then dst
  const int*   pairs = (const int*)d_in[2];     // [2,P]: u then v
  const float* w1    = (const float*)d_in[3];
  // d_in[4] conv1_b, d_in[6] conv2_b: constant column shifts cancel through
  // training-mode BatchNorm -> intentionally unused.
  const float* w2    = (const float*)d_in[5];
  const float* g1    = (const float*)d_in[7];
  const float* be1   = (const float*)d_in[8];
  const float* g2    = (const float*)d_in[9];
  const float* be2   = (const float*)d_in[10];
  const float* mw1   = (const float*)d_in[11];
  const float* mb1   = (const float*)d_in[12];
  const float* mw2   = (const float*)d_in[13];
  const float* mb2   = (const float*)d_in[14];
  const float* mw3   = (const float*)d_in[15];
  const float* mb3   = (const float*)d_in[16];

  char* base = (char*)d_ws;
  size_t off = 0;
  auto take = [&](size_t bytes) -> void* {
    void* p = base + off;
    off += (bytes + 255) & ~(size_t)255;
    return p;
  };
  float* dinv  = (float*)take((size_t)N_NODESC * 4);
  float* sums  = (float*)take(HIDC * 4);
  float* sumsq = (float*)take(HIDC * 4);
  float* ss    = (float*)take(2 * HIDC * 4);
  bf16*  w1b   = (bf16*)take((size_t)IN_DIMC * HIDC * 2);     // [N=256][K=128]
  bf16*  w2b   = (bf16*)take((size_t)HIDC * HIDC * 2);        // [256][256]
  bf16*  mw1b  = (bf16*)take((size_t)PAIR_DIMC * HIDC * 2);   // [256][768]
  bf16*  mw2b  = (bf16*)take((size_t)HIDC * 128 * 2);         // [128][256]
  bf16*  xb    = (bf16*)take((size_t)N_NODESC * IN_DIMC * 2);
  float* gout  = (float*)take((size_t)N_NODESC * HIDC * 4);
  bf16*  hb    = (bf16*)take((size_t)N_NODESC * HIDC * 2);    // contiguous after gout
  float* agg   = (float*)take((size_t)N_NODESC * HIDC * 4);
  bf16*  z1    = (bf16*)gout;   // reuse gout+hb region (153.6MB >= 134.2MB)
  bf16*  z2    = (bf16*)agg;    // reuse agg region (102.4MB >= 67.1MB)

  const int NT = 256;
  auto cdiv = [](int a, int b) { return (a + b - 1) / b; };

  // degrees -> dinv
  hipMemsetAsync(dinv, 0, (size_t)N_NODESC * 4, stream);
  k_deg<<<cdiv(N_EDGESC, NT), NT, 0, stream>>>(ei + N_EDGESC, dinv, N_EDGESC);
  k_dinv<<<cdiv(N_NODESC, NT), NT, 0, stream>>>(dinv, N_NODESC);

  // bf16 conversions (weights transposed to fragment-friendly [N][K])
  k_f32_to_bf16<<<cdiv(N_NODESC * IN_DIMC, NT), NT, 0, stream>>>(x, xb, N_NODESC * IN_DIMC);
  k_w_to_bf16t<<<cdiv(IN_DIMC * HIDC, NT), NT, 0, stream>>>(w1, w1b, IN_DIMC, HIDC);
  k_w_to_bf16t<<<cdiv(HIDC * HIDC, NT), NT, 0, stream>>>(w2, w2b, HIDC, HIDC);
  k_w_to_bf16t<<<cdiv(PAIR_DIMC * HIDC, NT), NT, 0, stream>>>(mw1, mw1b, PAIR_DIMC, HIDC);
  k_w_to_bf16t<<<cdiv(HIDC * 128, NT), NT, 0, stream>>>(mw2, mw2b, HIDC, 128);

  // ---- layer 1: XW1 -> aggregate -> BN+ReLU (bf16) ----
  dim3 gconv(HIDC / BN, cdiv(N_NODESC, BM));
  k_gemm<<<gconv, NT, 0, stream>>>(xb, w1b, gout, nullptr, nullptr, 0,
                                   N_NODESC, HIDC, IN_DIMC);
  hipMemsetAsync(agg, 0, (size_t)N_NODESC * HIDC * 4, stream);
  k_edge_scatter<<<cdiv(N_EDGESC * 64, NT), NT, 0, stream>>>(ei, ei + N_EDGESC, dinv, gout, agg, N_EDGESC);
  k_self_add<<<cdiv(N_NODESC * 64, NT), NT, 0, stream>>>(dinv, gout, agg, N_NODESC * 64);
  hipMemsetAsync(sums, 0, HIDC * 4, stream);
  hipMemsetAsync(sumsq, 0, HIDC * 4, stream);
  const int RPB = 256;
  k_colsum<<<cdiv(N_NODESC, RPB), HIDC, 0, stream>>>(agg, sums, sumsq, N_NODESC, RPB);
  k_bn_finalize<<<1, HIDC, 0, stream>>>(sums, sumsq, g1, be1, ss, N_NODESC);
  k_bn_relu_bf16<<<cdiv(N_NODESC * HIDC, NT), NT, 0, stream>>>(agg, ss, hb, N_NODESC * HIDC);

  // ---- layer 2: HW2 -> aggregate -> BN (f32, in place) ----
  k_gemm<<<gconv, NT, 0, stream>>>(hb, w2b, gout, nullptr, nullptr, 0,
                                   N_NODESC, HIDC, HIDC);
  hipMemsetAsync(agg, 0, (size_t)N_NODESC * HIDC * 4, stream);
  k_edge_scatter<<<cdiv(N_EDGESC * 64, NT), NT, 0, stream>>>(ei, ei + N_EDGESC, dinv, gout, agg, N_EDGESC);
  k_self_add<<<cdiv(N_NODESC * 64, NT), NT, 0, stream>>>(dinv, gout, agg, N_NODESC * 64);
  hipMemsetAsync(sums, 0, HIDC * 4, stream);
  hipMemsetAsync(sumsq, 0, HIDC * 4, stream);
  k_colsum<<<cdiv(N_NODESC, RPB), HIDC, 0, stream>>>(agg, sums, sumsq, N_NODESC, RPB);
  k_bn_finalize<<<1, HIDC, 0, stream>>>(sums, sumsq, g2, be2, ss, N_NODESC);
  k_bn_f32<<<cdiv(N_NODESC * HIDC, NT), NT, 0, stream>>>(agg, ss, N_NODESC * HIDC);

  // ---- decoder MLP ----
  dim3 gm1(HIDC / BN, N_PAIRSC / BM);
  k_gemm_feat<<<gm1, NT, 0, stream>>>(pairs, agg, x, mw1b, z1, mb1);
  dim3 gm2(128 / BN, N_PAIRSC / BM);
  k_gemm<<<gm2, NT, 0, stream>>>(z1, mw2b, nullptr, z2, mb2, 1,
                                 N_PAIRSC, 128, HIDC);
  k_mlp3<<<cdiv(N_PAIRSC, NT), NT, 0, stream>>>(z2, mw3, mb3, (float*)d_out, N_PAIRSC);
}
